// xLSTMModule_41154376630683
// MI455X (gfx1250) — matrix operand order repaired and checked by reading, not compiled, not run on hardware
//
#include <hip/hip_runtime.h>
#include <hip/hip_bf16.h>
#include <math.h>

#define Tn 64
#define Bn 8
#define Hn 128
#define NHn 4
#define Dn 256
#define EPSf 1e-5f

typedef __attribute__((ext_vector_type(16))) __bf16 v16bf;
typedef __attribute__((ext_vector_type(8)))  float  v8f;

// ---- workspace layout (float-element offsets) ----
#define OFF_XIN   (size_t)0          // [T,B,D]
#define OFF_TRIG  (size_t)131072     // [T,B,D]
#define OFF_QK    (size_t)262144     // [T,NH,B,D]
#define OFF_Q     (size_t)786432     // [T,NH,B,D]
#define OFF_K     (size_t)1310720
#define OFF_V     (size_t)1835008
#define OFF_OG    (size_t)2359296
#define OFF_SKIP  (size_t)2883584
#define OFF_H     (size_t)3407872
#define OFF_IT    (size_t)3932160    // [T,NH,B]
#define OFF_FT    (size_t)3934208
#define OFF_PK    (size_t)3936256    // bf16 packed weights: 5 mats x 4 heads x 65536

#define CSTRIDE 260                  // c row stride in floats (16B-aligned, padded)

__device__ __forceinline__ float sigf(float x){ return 1.f/(1.f+__expf(-x)); }
__device__ __forceinline__ float siluf(float x){ return x*sigf(x); }

// ---- pack Wq/Wk/Wv/Wo/Wskip (f32 [NH,256,256]) into bf16 WMMA B-fragment layout ----
// B (32x16 bf16): lane L<16: K0=kstep*32, n=ntile*16+L; lane>=16: K0=kstep*32+16, n=ntile*16+L-16
// element e (0..15) -> K = K0 + e.  Flat index f = ((nt*8+ks)*32+lane)*16+e.
__global__ __launch_bounds__(256) void kp_pack(
    const float* __restrict__ Wq, const float* __restrict__ Wk,
    const float* __restrict__ Wv, const float* __restrict__ Wo,
    const float* __restrict__ Wsk, float* __restrict__ ws)
{
  const int tid = threadIdx.x, m = blockIdx.z, h = blockIdx.y;
  const int f = blockIdx.x*256 + tid;
  const int e = f & 15, lane = (f>>4)&31, ks = (f>>9)&7, nt = f>>12;
  const int K = ks*32 + ((lane & 16) ? 16 : 0) + e;
  const int n = nt*16 + (lane & 15);
  const float* W = (m==0)?Wq:(m==1)?Wk:(m==2)?Wv:(m==3)?Wo:Wsk;
  __bf16* dst = (__bf16*)(ws + OFF_PK);
  dst[(size_t)(m*NHn + h)*65536 + f] = (__bf16)W[(size_t)h*65536 + (size_t)K*Dn + n];
}

// ---- per-timestep preprocessing: BN1, up-projections, conv+silu, it/ft ----
__global__ __launch_bounds__(256) void k0_pre(
    const float* __restrict__ x,
    const float* __restrict__ Wup1, const float* __restrict__ bup1,
    const float* __restrict__ Wup2, const float* __restrict__ bup2,
    const float* __restrict__ bn1s, const float* __restrict__ bn1b,
    const float* __restrict__ ck4,  const float* __restrict__ cb4,
    const float* __restrict__ Wi,   const float* __restrict__ biv,
    const float* __restrict__ Wf,   const float* __restrict__ bfv,
    float* __restrict__ ws)
{
  __shared__ float su[Bn*Hn];   // BN1 output u
  __shared__ float sx[Bn*Dn];   // xin
  const int t = blockIdx.x, tid = threadIdx.x;
  float* xing = ws + OFF_XIN;
  float* trg  = ws + OFF_TRIG;
  float* qkg  = ws + OFF_QK;
  float* itg  = ws + OFF_IT;
  float* ftg  = ws + OFF_FT;

  if (tid < Hn) {
    float vals[Bn]; float mu=0.f, m2=0.f;
    #pragma unroll
    for (int b=0;b<Bn;++b){ float v = x[(size_t)t*Bn*Hn + b*Hn + tid]; vals[b]=v; mu+=v; m2+=v*v; }
    mu *= 0.125f; float var = m2*0.125f - mu*mu;
    float sc = rsqrtf(var+EPSf)*bn1s[tid]; float bb = bn1b[tid];
    #pragma unroll
    for (int b=0;b<Bn;++b) su[b*Hn+tid] = (vals[b]-mu)*sc + bb;
  }
  __syncthreads();

  { // xin = u@Wup1+bup1 ; trig = silu(u@Wup2+bup2)   (thread = output column j)
    const int j = tid;
    float o1[Bn], o2[Bn];
    #pragma unroll
    for (int b=0;b<Bn;++b){ o1[b]=0.f; o2[b]=0.f; }
    for (int kk=0; kk<Hn; ++kk) {
      float w1 = Wup1[kk*Dn + j];
      float w2 = Wup2[kk*Dn + j];
      #pragma unroll
      for (int b=0;b<Bn;++b){ float uv = su[b*Hn+kk]; o1[b]=fmaf(uv,w1,o1[b]); o2[b]=fmaf(uv,w2,o2[b]); }
    }
    #pragma unroll
    for (int b=0;b<Bn;++b){
      float xv = o1[b] + bup1[j];
      sx[b*Dn+j] = xv;
      xing[(size_t)t*Bn*Dn + b*Dn + j] = xv;
      float tv = o2[b] + bup2[j];
      trg[(size_t)t*Bn*Dn + b*Dn + j] = siluf(tv);
    }
  }
  __syncthreads();

  { // depthwise feature conv k=4, SAME (pad_lo=1), then silu -> qk[t,h,b,:]
    const int j = tid;
    float ck[4][NHn];
    #pragma unroll
    for (int i=0;i<4;++i)
      #pragma unroll
      for (int h=0;h<NHn;++h) ck[i][h] = ck4[i*NHn + h];
    #pragma unroll
    for (int h=0;h<NHn;++h){
      float cb = cb4[h];
      for (int b=0;b<Bn;++b){
        float s = cb;
        #pragma unroll
        for (int i=0;i<4;++i){
          int p = j - 1 + i;
          if (p >= 0 && p < Dn) s = fmaf(sx[b*Dn + p], ck[i][h], s);
        }
        qkg[(((size_t)t*NHn + h)*Bn + b)*Dn + j] = siluf(s);
      }
    }
  }

  { // it/ft: 32 dot products of length 256; 8 waves x 4 pairs, wave-reduce
    const int lane = tid & 31, wave = tid >> 5;
    #pragma unroll
    for (int pp=0; pp<4; ++pp) {
      int p = wave*4 + pp;
      int h = p >> 3, b = p & 7;
      float si = 0.f, sf = 0.f;
      #pragma unroll
      for (int s8=0; s8<8; ++s8){
        int d2 = lane + 32*s8;
        float xv = sx[b*Dn + d2];
        si = fmaf(xv, Wi[h*Dn + d2], si);
        sf = fmaf(xv, Wf[h*Dn + d2], sf);
      }
      #pragma unroll
      for (int off=16; off>0; off>>=1){ si += __shfl_xor(si, off); sf += __shfl_xor(sf, off); }
      if (lane == 0){
        itg[t*32 + p] = si + biv[h];
        ftg[t*32 + p] = sf + bfv[h];
      }
    }
  }
}

// ---- per (t,head) WMMA GEMMs: q,k,v,outg,skip = A[8(pad16),256] @ W[256,256] ----
__global__ __launch_bounds__(128) void k1_gemms(
    const float* __restrict__ bq, const float* __restrict__ bk,
    const float* __restrict__ bv, const float* __restrict__ bo,
    const float* __restrict__ bsk, float* __restrict__ ws)
{
  __shared__ __bf16 sAqk[4096];   // A-fragment layout, 8 ksteps x 32 lanes x 16
  __shared__ __bf16 sAxin[4096];
  const int t = blockIdx.x, h = blockIdx.y, tid = threadIdx.x;
  const float* qkg  = ws + OFF_QK;
  const float* xing = ws + OFF_XIN;
  const __bf16* pk = (const __bf16*)(ws + OFF_PK);

  // pack activations into bf16 A-fragment layout in LDS
  // A (16x32 bf16): lane L<16: M=L, K = kstep*32 + e; lane>=16: +16 to K
  for (int f = tid; f < 4096; f += 128) {
    int e = f & 15, lane = (f>>4)&31, ks = f>>9;
    int K = ks*32 + ((lane & 16) ? 16 : 0) + e;
    int M = lane & 15;
    sAqk[f]  = (M<Bn) ? (__bf16)qkg[(((size_t)t*NHn+h)*Bn+M)*Dn + K] : (__bf16)0.f;
    sAxin[f] = (M<Bn) ? (__bf16)xing[((size_t)t*Bn+M)*Dn + K] : (__bf16)0.f;
  }
  __syncthreads();

  const int lane = tid & 31, wave = tid >> 5;

  #pragma unroll
  for (int g=0; g<5; ++g) {                  // 0=q 1=k 2=v 3=outg 4=skip
    const __bf16* sA = (g==2 || g==3) ? sAxin : sAqk;
    const __bf16* pw = pk + (size_t)(g*NHn + h)*65536;
    v8f acc[4] = {};
    for (int ks=0; ks<8; ++ks) {
      v16bf a = *(const v16bf*)(sA + ks*512 + lane*16);
      #pragma unroll
      for (int jj=0; jj<4; ++jj) {
        int nt = wave*4 + jj;
        v16bf bb = *(const v16bf*)(pw + ((size_t)(nt*8+ks)*32 + lane)*16);
        acc[jj] = __builtin_amdgcn_wmma_f32_16x16x32_bf16(
            false, a, false, bb, (short)0, acc[jj], false, false);
      }
    }
    float* outp; const float* bp;
    switch (g) {
      case 0:  outp = ws+OFF_Q;    bp = bq;  break;
      case 1:  outp = ws+OFF_K;    bp = bk;  break;
      case 2:  outp = ws+OFF_V;    bp = bv;  break;
      case 3:  outp = ws+OFF_OG;   bp = bo;  break;
      default: outp = ws+OFF_SKIP; bp = bsk; break;
    }
    if (lane < 16) {  // rows M=0..7 live in lanes 0..15, VGPR r -> M=r
      #pragma unroll
      for (int jj=0;jj<4;++jj){
        int n = (wave*4+jj)*16 + lane;
        float bia = bp[h*Dn + n];
        #pragma unroll
        for (int r=0;r<Bn;++r){
          float val = acc[jj][r] + bia;
          if (g==1) val *= 0.0625f;          // * 1/sqrt(D)
          if (g==3) val = sigf(val);         // output gate
          outp[(((size_t)t*NHn+h)*Bn + r)*Dn + n] = val;
        }
      }
    }
  }
}

// ---- sequential mLSTM recurrence: 32 blocks (one per (h,b)), c state LDS-resident ----
// c stored with row stride 260 floats (16B aligned) -> float4 LDS traffic.
__global__ __launch_bounds__(256) void k2_recur(float* __restrict__ ws)
{
  extern __shared__ float sm[];
  const int hb = blockIdx.x, tid = threadIdx.x, lane = tid & 31, wave = tid >> 5;
  float* cm  = sm;                      // c: 256 rows x stride 260 = 66560 floats
  float* sq  = sm + 66560;
  float* sk  = sq + Dn;
  float* sv  = sk + Dn;
  float* sog = sv + Dn;
  float* ssk = sog + Dn;
  float* fgl = ssk + Dn;                // 64
  float* igl = fgl + Tn;                // 64
  float* red = igl + Tn;                // 8
  float* dv  = red + 8;                 // 1
  const float* qg  = ws + OFF_Q;
  const float* kg  = ws + OFF_K;
  const float* vg  = ws + OFF_V;
  const float* ogg = ws + OFF_OG;
  const float* skg = ws + OFF_SKIP;
  const float* itg = ws + OFF_IT;
  const float* ftg = ws + OFF_FT;
  float* hg = ws + OFF_H;

  for (int idx = tid; idx < 256*CSTRIDE; idx += 256) cm[idx] = 0.f;
  if (tid == 0) {           // scalar m-max scan -> fg/ig for all t
    float m = 0.f;
    for (int t=0;t<Tn;++t){
      float iv = itg[t*32 + hb], fv = ftg[t*32 + hb];
      float mn = fmaxf(fv + m, iv);
      igl[t] = __expf(iv - mn);
      fgl[t] = __expf(fv + m - mn);
      m = mn;
    }
  }
  float4* c4 = (float4*)(cm + tid*CSTRIDE);       // thread tid owns row i=tid of c
  const float4* k4 = (const float4*)sk;
  const float4* q4 = (const float4*)sq;

  float ni = 0.f;           // thread tid owns n[tid]
  for (int t=0;t<Tn;++t) {
    __syncthreads();
    const size_t base = ((size_t)t*32 + hb)*Dn + tid;
    sq[tid]=qg[base]; sk[tid]=kg[base]; sv[tid]=vg[base]; sog[tid]=ogg[base]; ssk[tid]=skg[base];
    if (t+1 < Tn) {         // global_prefetch_b8 next step's vectors
      __builtin_prefetch(qg  + base + 8192, 0, 1);
      __builtin_prefetch(kg  + base + 8192, 0, 1);
      __builtin_prefetch(vg  + base + 8192, 0, 1);
      __builtin_prefetch(ogg + base + 8192, 0, 1);
      __builtin_prefetch(skg + base + 8192, 0, 1);
    }
    __syncthreads();
    const float fg = fgl[t], ig = igl[t];
    const float igv = ig * sv[tid];
    float acc = 0.f;
    #pragma unroll 4
    for (int kk=0; kk<Dn/4; ++kk) {
      float4 cv = c4[kk];
      const float4 kv = k4[kk];                   // broadcast, conflict-free
      const float4 qv = q4[kk];
      cv.x = fmaf(fg, cv.x, igv * kv.x);          // c = fg*c + ig*v_i*k
      cv.y = fmaf(fg, cv.y, igv * kv.y);
      cv.z = fmaf(fg, cv.z, igv * kv.z);
      cv.w = fmaf(fg, cv.w, igv * kv.w);
      c4[kk] = cv;
      acc = fmaf(cv.x, qv.x, acc);                // (c@q)_i
      acc = fmaf(cv.y, qv.y, acc);
      acc = fmaf(cv.z, qv.z, acc);
      acc = fmaf(cv.w, qv.w, acc);
    }
    ni = fmaf(fg, ni, ig * sk[tid]);              // n = fg*n + ig*k
    float p = ni * sq[tid];                       // partial of n.q
    #pragma unroll
    for (int off=16; off>0; off>>=1) p += __shfl_xor(p, off);
    if (lane == 0) red[wave] = p;
    __syncthreads();
    if (tid == 0){ float s=0.f; for (int w2=0;w2<8;++w2) s += red[w2]; dv[0] = s; }
    __syncthreads();
    const float scaler = 1.f / fmaxf(fabsf(dv[0]), 1.f);
    hg[base] = fmaf(sog[tid]*acc, scaler, ssk[tid]);
  }
}

// ---- per-timestep epilogue: BN2, Wpoll, *trig, Wdown, +residual ----
__global__ __launch_bounds__(256) void k3_out(
    const float* __restrict__ x,
    const float* __restrict__ bn2s, const float* __restrict__ bn2b,
    const float* __restrict__ Wpoll, const float* __restrict__ bpoll,
    const float* __restrict__ Wdown, const float* __restrict__ bdown,
    const float* __restrict__ ws, float* __restrict__ out)
{
  __shared__ float sh[NHn*Bn*Dn];   // 8192
  __shared__ float strig[Bn*Dn];    // 2048
  __shared__ float gl[Bn*Dn];       // 2048
  const int t = blockIdx.x, tid = threadIdx.x;
  const float* hg  = ws + OFF_H;
  const float* trg = ws + OFF_TRIG;
  for (int idx = tid; idx < NHn*Bn*Dn; idx += 256) sh[idx] = hg[(size_t)t*8192 + idx];
  for (int idx = tid; idx < Bn*Dn;    idx += 256) strig[idx] = trg[(size_t)t*2048 + idx];
  __syncthreads();
  { // BN2: stats over 32 (nh,b) per feature d=tid
    const int d = tid;
    float mu=0.f, m2=0.f;
    #pragma unroll
    for (int hb=0; hb<32; ++hb){ float v = sh[hb*Dn + d]; mu+=v; m2+=v*v; }
    mu *= (1.f/32.f); float var = m2*(1.f/32.f) - mu*mu;
    float sc = rsqrtf(var+EPSf)*bn2s[d]; float bb = bn2b[d];
    #pragma unroll
    for (int hb=0; hb<32; ++hb) sh[hb*Dn + d] = (sh[hb*Dn + d]-mu)*sc + bb;
  }
  __syncthreads();
  { // poll: [8,1024]@[1024,256], thread = output column n
    const int n = tid;
    float accb[Bn];
    #pragma unroll
    for (int b=0;b<Bn;++b) accb[b]=0.f;
    for (int m=0;m<NHn*Dn;++m){
      float w = Wpoll[(size_t)m*Dn + n];
      int hh = m >> 8, dd = m & 255;         // flat m = h*D + d
      #pragma unroll
      for (int b=0;b<Bn;++b) accb[b] = fmaf(sh[(hh*Bn+b)*Dn + dd], w, accb[b]);
    }
    float bpv = bpoll[n];
    #pragma unroll
    for (int b=0;b<Bn;++b) gl[b*Dn + n] = (accb[b] + bpv) * strig[b*Dn + n];
  }
  __syncthreads();
  { // down: [8,256]@[256,128] + bdown + residual
    const int o = tid & 127, grp = tid >> 7;
    #pragma unroll
    for (int bi4=0; bi4<4; ++bi4){
      int b = grp*4 + bi4;
      float acc = 0.f;
      for (int n=0;n<Dn;++n) acc = fmaf(gl[b*Dn + n], Wdown[n*Hn + o], acc);
      out[(size_t)t*Bn*Hn + b*Hn + o] = acc + bdown[o] + x[(size_t)t*Bn*Hn + b*Hn + o];
    }
  }
}

extern "C" void kernel_launch(void* const* d_in, const int* in_sizes, int n_in,
                              void* d_out, int out_size, void* d_ws, size_t ws_size,
                              hipStream_t stream)
{
  const float* x     = (const float*)d_in[0];
  const float* Wq    = (const float*)d_in[1];
  const float* bq    = (const float*)d_in[2];
  const float* Wk    = (const float*)d_in[3];
  const float* bk    = (const float*)d_in[4];
  const float* Wv    = (const float*)d_in[5];
  const float* bv    = (const float*)d_in[6];
  const float* convk = (const float*)d_in[7];
  const float* convb = (const float*)d_in[8];
  const float* Wi    = (const float*)d_in[9];
  const float* biv   = (const float*)d_in[10];
  const float* Wf    = (const float*)d_in[11];
  const float* bfv   = (const float*)d_in[12];
  const float* Wo    = (const float*)d_in[13];
  const float* bo    = (const float*)d_in[14];
  const float* Wsk   = (const float*)d_in[15];
  const float* bsk   = (const float*)d_in[16];
  const float* bn1s  = (const float*)d_in[17];
  const float* bn1b  = (const float*)d_in[18];
  const float* bn2s  = (const float*)d_in[19];
  const float* bn2b  = (const float*)d_in[20];
  const float* Wup1  = (const float*)d_in[21];
  const float* bup1  = (const float*)d_in[22];
  const float* Wup2  = (const float*)d_in[23];
  const float* bup2  = (const float*)d_in[24];
  const float* Wpoll = (const float*)d_in[25];
  const float* bpoll = (const float*)d_in[26];
  const float* Wdown = (const float*)d_in[27];
  const float* bdown = (const float*)d_in[28];
  float* ws  = (float*)d_ws;
  float* out = (float*)d_out;

  kp_pack<<<dim3(256, NHn, 5), 256, 0, stream>>>(Wq, Wk, Wv, Wo, Wsk, ws);
  k0_pre<<<Tn, 256, 0, stream>>>(x, Wup1, bup1, Wup2, bup2, bn1s, bn1b,
                                 convk, convb, Wi, biv, Wf, bfv, ws);
  k1_gemms<<<dim3(Tn, NHn), 128, 0, stream>>>(bq, bk, bv, bo, bsk, ws);
  constexpr size_t K2_SHM = (size_t)(256*CSTRIDE + 5*256 + 2*64 + 8 + 1) * sizeof(float); // ~272KB
  k2_recur<<<32, 256, K2_SHM, stream>>>(ws);
  k3_out<<<Tn, 256, 0, stream>>>(x, bn2s, bn2b, Wpoll, bpoll, Wdown, bdown, ws, out);
}
